// NetVLAD_22479858827339
// MI455X (gfx1250) — matrix operand order repaired and checked by reading, not compiled, not run on hardware
//
#include <hip/hip_runtime.h>

typedef __attribute__((ext_vector_type(16))) __bf16        v16bf;
typedef __attribute__((ext_vector_type(8)))  float         v8f;
typedef __attribute__((ext_vector_type(8)))  unsigned int  v8u;
typedef __attribute__((ext_vector_type(4)))  unsigned int  v4u;
typedef __attribute__((ext_vector_type(2)))  unsigned int  v2u;

#define C_DIM   512
#define K_DIM   64
#define S_DIM   4096
#define N_IMG   16
#define TILE_S  128
#define CHUNK_TILES 8
#define CHUNK_S (TILE_S*CHUNK_TILES)   // 1024
#define NCHUNK  (S_DIM/CHUNK_S)        // 4

// padded LDS row strides (ushort units): +8 shifts consecutive rows by 4 banks
#define XS_LD   (TILE_S + 8)           // xs: [C_DIM][XS_LD]   s-innermost
#define XT_LD   (C_DIM  + 8)           // xT: [TILE_S][XT_LD]  c-innermost
#define AH_LD   (TILE_S + 8)           // ah: [K_DIM][AH_LD]   s-innermost

#define SMEM_BYTES ((C_DIM*XS_LD + TILE_S*XT_LD + K_DIM*AH_LD)*2 + (TILE_S + 256 + K_DIM)*4)

__device__ __forceinline__ unsigned short f2bfu(float f) {
    unsigned int u = __float_as_uint(f);
    u += 0x7FFFu + ((u >> 16) & 1u);          // round-to-nearest-even
    return (unsigned short)(u >> 16);
}

// two 16-byte loads -> one bf16 fragment (lowers to 2x ds_load_b128 / global_load_b128)
struct u32x8 { v4u lo, hi; };
__device__ __forceinline__ v16bf ld_frag(const unsigned int* p0, const unsigned int* p1) {
    u32x8 t;
    t.lo = *(const v4u*)p0;
    t.hi = *(const v4u*)p1;
    return __builtin_bit_cast(v16bf, t);
}

// one-time conv_w f32 -> bf16 into workspace (stays resident in L2/WGP$)
__global__ void __launch_bounds__(256)
netvlad_wcvt(const float* __restrict__ w, unsigned short* __restrict__ wb)
{
    int i = blockIdx.x * 256 + threadIdx.x;
    wb[i] = f2bfu(w[i]);
}

__global__ void __launch_bounds__(256, 1)
netvlad_main(const float* __restrict__ x, const unsigned short* __restrict__ wb,
             float* __restrict__ agg_ws, float* __restrict__ asum_ws)
{
    extern __shared__ char smem[];
    unsigned short* xs  = (unsigned short*)smem;          // [512][XS_LD] raw x, bf16 (GEMM2 B)
    unsigned short* xT  = xs + C_DIM*XS_LD;               // [128][XT_LD] raw x^T, bf16 (GEMM1 B)
    unsigned short* ah  = xT + TILE_S*XT_LD;              // [64][AH_LD]  a*inv, bf16 (GEMM2 A)
    float* inv  = (float*)(ah + K_DIM*AH_LD);             // [128] 1/max(||x_p||,eps)
    float* psum = inv + TILE_S;                           // [256] partial sumsq
    float* asum = psum + 256;                             // [64]  sum_s a[k,s]

    const int tid     = threadIdx.x;
    const int wave    = tid >> 5;
    const int lane    = tid & 31;
    const int l16     = lane & 15;
    const int halfsel = lane >> 4;                        // 0: lanes 0-15, 1: lanes 16-31
    const int n       = blockIdx.y;
    const int chunk   = blockIdx.x;

    if (tid < K_DIM) asum[tid] = 0.f;

    // GEMM2 accumulators: wave owns c in [wave*64, wave*64+64), all 64 k
    v8f acc[4][4];
    #pragma unroll
    for (int rk = 0; rk < 4; ++rk)
        #pragma unroll
        for (int ct = 0; ct < 4; ++ct)
            #pragma unroll
            for (int j = 0; j < 8; ++j) acc[rk][ct][j] = 0.f;

    // register accumulators for sum_s a[k,s] (lane-local, reduced once at end)
    v8f aS[4];
    #pragma unroll
    for (int r = 0; r < 4; ++r)
        #pragma unroll
        for (int j = 0; j < 8; ++j) aS[r][j] = 0.f;

    const float* xbase = x + (size_t)n * C_DIM * S_DIM;
    const int pbase = wave * 16;

    __syncthreads();

    for (int t = 0; t < CHUNK_TILES; ++t) {
        const int s0 = chunk*CHUNK_S + t*TILE_S;

        // launder a zero offset: defeats LICM of the (t-invariant) W fragment
        // loads out of the tile loop (prevents the 2KB/lane spill), while the
        // address stays derived from the kernarg pointer so address-space
        // inference keeps these as global_load_b128 (not flat_load).
        unsigned int woff = 0;
        asm volatile("" : "+s"(woff));
        const unsigned int* wt = (const unsigned int*)wb + woff;

        // ---- load x tile (coalesced), write BOTH layouts, per-pixel sumsq ----
        {
            const int p   = tid & (TILE_S - 1);
            const int ch0 = (tid >> 7) * (C_DIM/2);
            const float* xp = xbase + s0 + p;
            float ss = 0.f;
            for (int c = 0; c < C_DIM/2; c += 4) {
                float v0 = xp[(size_t)(ch0 + c    ) * S_DIM];
                float v1 = xp[(size_t)(ch0 + c + 1) * S_DIM];
                float v2 = xp[(size_t)(ch0 + c + 2) * S_DIM];
                float v3 = xp[(size_t)(ch0 + c + 3) * S_DIM];
                ss += v0*v0 + v1*v1 + v2*v2 + v3*v3;
                unsigned int b0 = f2bfu(v0), b1 = f2bfu(v1);
                unsigned int b2 = f2bfu(v2), b3 = f2bfu(v3);
                xs[(ch0 + c    )*XS_LD + p] = (unsigned short)b0;
                xs[(ch0 + c + 1)*XS_LD + p] = (unsigned short)b1;
                xs[(ch0 + c + 2)*XS_LD + p] = (unsigned short)b2;
                xs[(ch0 + c + 3)*XS_LD + p] = (unsigned short)b3;
                v2u pk; pk[0] = b0 | (b1 << 16); pk[1] = b2 | (b3 << 16);
                *(v2u*)&xT[p*XT_LD + ch0 + c] = pk;        // ds_store_b64
            }
            psum[tid] = ss;
        }

        // ---- prefetch next tile into caches while this tile's GEMMs run ----
        // 2048 x 128B lines cover the next 256KB tile; bijective over 256 thr x 8
        if (t + 1 < CHUNK_TILES) {
            const float* xn = xbase + s0 + TILE_S;
            #pragma unroll
            for (int i = 0; i < 8; ++i) {
                int line = tid + i*256;            // 0..2047
                int c    = line >> 2;              // channel row 0..511
                int pl   = (line & 3) << 5;        // 0,32,64,96 floats into the row
                __builtin_prefetch(xn + (size_t)c*S_DIM + pl, 0, 3);
            }
        }

        __syncthreads();
        if (tid < TILE_S) {
            float nrm = sqrtf(psum[tid] + psum[tid + TILE_S]);
            inv[tid]  = 1.0f / fmaxf(nrm, 1e-12f);
        }
        __syncthreads();

        // ---- GEMM1: logits[k, p]; wave owns p-tile = wave*16, all 4 k-tiles ----
        v8f L[4];
        #pragma unroll
        for (int r = 0; r < 4; ++r)
            #pragma unroll
            for (int j = 0; j < 8; ++j) L[r][j] = 0.f;

        const unsigned int* xtrow = (const unsigned int*)(xT + (pbase + l16)*XT_LD);

        #pragma unroll 2
        for (int kk = 0; kk < C_DIM/32; ++kk) {
            // B fragment (32x16, K=c, N=p): dwords kk*16 + 4*half + [0..3], +8 + [0..3]  (2x ds_load_b128)
            const unsigned int* bp = xtrow + kk*16 + halfsel*4;
            v16bf bfrag = ld_frag(bp, bp + 8);
            #pragma unroll
            for (int r = 0; r < 4; ++r) {
                // A fragment (16x32, M=k, K=c) from bf16 W in global (2x b128, L2-resident)
                const unsigned int* ap = wt + (r*16 + l16)*(C_DIM/2) + kk*16 + halfsel*4;
                v16bf afrag = ld_frag(ap, ap + 8);
                L[r] = __builtin_amdgcn_wmma_f32_16x16x32_bf16(
                           false, afrag, false, bfrag, (short)0, L[r], false, false);
            }
        }

        // ---- fused normalize + softmax over k (registers + one wave32 shuffle) ----
        {
            const int   p    = pbase + l16;
            const float invp = inv[p];
            float mx = -1e30f;
            #pragma unroll
            for (int r = 0; r < 4; ++r)
                #pragma unroll
                for (int j = 0; j < 8; ++j) { L[r][j] *= invp; mx = fmaxf(mx, L[r][j]); }
            mx = fmaxf(mx, __shfl_xor(mx, 16));
            float sum = 0.f;
            #pragma unroll
            for (int r = 0; r < 4; ++r)
                #pragma unroll
                for (int j = 0; j < 8; ++j) { float e = __expf(L[r][j] - mx); L[r][j] = e; sum += e; }
            sum += __shfl_xor(sum, 16);
            const float rs = 1.0f / sum;
            #pragma unroll
            for (int r = 0; r < 4; ++r)
                #pragma unroll
                for (int j = 0; j < 8; ++j) {
                    float a = L[r][j] * rs;                 // softmax weight
                    int   k = r*16 + j + halfsel*8;         // D layout: M=vgpr(+8 hi lanes)
                    aS[r][j] += a;                          // register asum accumulation
                    ah[k*AH_LD + p] = f2bfu(a * invp);      // fold 1/||x|| into a for GEMM2
                }
        }
        __syncthreads();

        // ---- GEMM2: agg[k, c] += a_hat(64x128) @ x_tile^T(128x512) ----
        for (int ss = 0; ss < TILE_S/32; ++ss) {
            v16bf afr[4], bfr[4];
            #pragma unroll
            for (int rk = 0; rk < 4; ++rk) {
                const unsigned int* ap =
                    (const unsigned int*)(ah + (rk*16 + l16)*AH_LD) + ss*16 + halfsel*4;
                afr[rk] = ld_frag(ap, ap + 8);
            }
            #pragma unroll
            for (int ct = 0; ct < 4; ++ct) {
                const unsigned int* bp =
                    (const unsigned int*)(xs + (wave*64 + ct*16 + l16)*XS_LD) + ss*16 + halfsel*4;
                bfr[ct] = ld_frag(bp, bp + 8);
            }
            #pragma unroll
            for (int rk = 0; rk < 4; ++rk)
                #pragma unroll
                for (int ct = 0; ct < 4; ++ct)
                    acc[rk][ct] = __builtin_amdgcn_wmma_f32_16x16x32_bf16(
                        false, afr[rk], false, bfr[ct], (short)0, acc[rk][ct], false, false);
        }
        __syncthreads();   // before next tile overwrites xs / xT / ah
    }

    // ---- reduce register asum: butterfly over the 16 p-columns of this wave ----
    #pragma unroll
    for (int r = 0; r < 4; ++r)
        #pragma unroll
        for (int j = 0; j < 8; ++j) {
            float v = aS[r][j];
            v += __shfl_xor(v, 1);
            v += __shfl_xor(v, 2);
            v += __shfl_xor(v, 4);
            v += __shfl_xor(v, 8);
            if (l16 == 0)
                atomicAdd(&asum[r*16 + j + halfsel*8], v);   // 8 waves x 2 halves
        }

    // ---- write partials: agg[n][chunk][k][c], asum[n][chunk][k] ----
    float* aggp = agg_ws + ((size_t)(n*NCHUNK + chunk)) * K_DIM * C_DIM;
    #pragma unroll
    for (int rk = 0; rk < 4; ++rk)
        #pragma unroll
        for (int ct = 0; ct < 4; ++ct)
            #pragma unroll
            for (int j = 0; j < 8; ++j) {
                int k = rk*16 + j + halfsel*8;
                int c = wave*64 + ct*16 + l16;
                aggp[k*C_DIM + c] = acc[rk][ct][j];
            }
    __syncthreads();
    if (tid < K_DIM)
        asum_ws[((size_t)(n*NCHUNK + chunk))*K_DIM + tid] = asum[tid];
}

// vlad[n,k,c] = sum_chunk agg - centroids[k,c] * sum_chunk asum
__global__ void __launch_bounds__(256)
netvlad_combine(const float* __restrict__ agg_ws, const float* __restrict__ asum_ws,
                const float* __restrict__ centroids, float* __restrict__ out)
{
    int idx = blockIdx.x * 256 + threadIdx.x;          // over N*K*C = 524288
    int c = idx & (C_DIM - 1);
    int k = (idx >> 9) & (K_DIM - 1);
    int n = idx >> 15;
    float s = 0.f, t = 0.f;
    #pragma unroll
    for (int ch = 0; ch < NCHUNK; ++ch) {
        s += agg_ws[((size_t)(n*NCHUNK + ch))*K_DIM*C_DIM + k*C_DIM + c];
        t += asum_ws[((size_t)(n*NCHUNK + ch))*K_DIM + k];
    }
    out[idx] = s - centroids[k*C_DIM + c] * t;
}

extern "C" void kernel_launch(void* const* d_in, const int* in_sizes, int n_in,
                              void* d_out, int out_size, void* d_ws, size_t ws_size,
                              hipStream_t stream)
{
    (void)in_sizes; (void)n_in; (void)out_size; (void)ws_size;
    const float* x         = (const float*)d_in[0];
    const float* conv_w    = (const float*)d_in[1];
    const float* centroids = (const float*)d_in[2];
    float* out = (float*)d_out;

    float* agg_ws  = (float*)d_ws;                               // 16*4*64*512 f32 = 8 MB
    float* asum_ws = agg_ws + (size_t)N_IMG*NCHUNK*K_DIM*C_DIM;  // 16*4*64 f32
    unsigned short* wbf = (unsigned short*)(asum_ws + (size_t)N_IMG*NCHUNK*K_DIM); // 64KB

    netvlad_wcvt<<<(K_DIM*C_DIM)/256, 256, 0, stream>>>(conv_w, wbf);

    dim3 grid(NCHUNK, N_IMG);
    netvlad_main<<<grid, dim3(256), SMEM_BYTES, stream>>>(x, wbf, agg_ws, asum_ws);

    int total = N_IMG * K_DIM * C_DIM;
    netvlad_combine<<<total / 256, 256, 0, stream>>>(agg_ws, asum_ws, centroids, out);
}
